// FsModel_13280038879787
// MI455X (gfx1250) — matrix operand order repaired and checked by reading, not compiled, not run on hardware
//
#include <hip/hip_runtime.h>
#include <hip/hip_bf16.h>

typedef __attribute__((ext_vector_type(16))) __bf16          v16bf;
typedef __attribute__((ext_vector_type(8)))  float           v8f;
typedef __attribute__((ext_vector_type(8)))  unsigned short  u16x8;
typedef __attribute__((ext_vector_type(4)))  unsigned short  u16x4;
typedef int v4i32 __attribute__((vector_size(16)));

union bfvec { v16bf v; u16x8 h[2]; };

#if __has_builtin(__builtin_amdgcn_global_load_async_to_lds_b128) && \
    __has_builtin(__builtin_amdgcn_s_wait_asynccnt)
#define USE_ASYNC_LDS 1
#else
#define USE_ASYNC_LDS 0
#endif

__device__ __forceinline__ unsigned short f2bf(float f) {
    unsigned u = __float_as_uint(f);
    u += 0x7FFFu + ((u >> 16) & 1u);          // round-to-nearest-even
    return (unsigned short)(u >> 16);
}
__device__ __forceinline__ float bf2f(unsigned short h) {
    return __uint_as_float(((unsigned)h) << 16);
}

// ---------------------------------------------------------------- prep kernels
__global__ void cvt_f32_bf16(const float* __restrict__ in, unsigned short* __restrict__ out, long n) {
    long i = (long)blockIdx.x * blockDim.x + threadIdx.x;
    if (i < n) out[i] = f2bf(in[i]);
}

// OIHW f32 -> [tap][Cout][Cin] bf16
__global__ void pack_weights(const float* __restrict__ W, unsigned short* __restrict__ wr,
                             int Cout, int Cin) {
    long total = (long)Cout * Cin * 9;
    long i = (long)blockIdx.x * blockDim.x + threadIdx.x;
    if (i >= total) return;
    int tap = (int)(i / ((long)Cout * Cin));
    long rem = i % ((long)Cout * Cin);
    int co = (int)(rem / Cin), ci = (int)(rem % Cin);
    wr[i] = f2bf(W[((long)(co * Cin + ci)) * 9 + tap]);
}

__global__ void fold_bn(const float* __restrict__ b, const float* __restrict__ g,
                        const float* __restrict__ be, const float* __restrict__ mn,
                        const float* __restrict__ vr, float* __restrict__ scale,
                        float* __restrict__ bias, int C) {
    int c = blockIdx.x * blockDim.x + threadIdx.x;
    if (c >= C) return;
    float s = g[c] * rsqrtf(vr[c] + 1e-5f);
    scale[c] = s;
    bias[c]  = (b[c] - mn[c]) * s + be[c];
}

// ---------------------------------------------------------------- implicit-GEMM conv
// in : bf16 NCHW   wr : bf16 [tap][Cout][Cin]   out : bf16 NCHW (post scale/bias/relu)
// block = 256 threads (8 waves), tile 64(Cout) x 64(pixels), K-step 32, double-buffered LDS
__global__ __launch_bounds__(256) void conv_bn_relu_wmma(
        const unsigned short* __restrict__ in, const unsigned short* __restrict__ wr,
        const float* __restrict__ scale, const float* __restrict__ bias,
        unsigned short* __restrict__ out, int Cin, int Cout, int H, int W) {

    __shared__ __align__(16) unsigned short As[2][3][64][40];  // [buf][s-tap][m][k]
    __shared__ __align__(16) unsigned short Bs[2][80][40];     // [buf][pixel (x0-8..x0+71)][k]

    const int tid  = threadIdx.x;
    const int lane = tid & 31;
    const int wave = tid >> 5;
    const int wm   = wave >> 1;          // 4 waves along M
    const int wn   = wave & 1;           // 2 waves along N
    const int x0   = blockIdx.x * 64;
    const int y    = blockIdx.y;
    const int cg   = Cout >> 6;
    const int n    = blockIdx.z / cg;
    const int co0  = (blockIdx.z % cg) << 6;

    const int ksC = (Cin + 31) >> 5;     // k-steps per tap row
    const int K   = 3 * ksC;             // total k-steps (3 tap rows)

    const int am  = tid >> 2;            // A staging: row 0..63
    const int ach = tid & 3;             // A staging: 8-elem chunk 0..3

    v8f acc0 = {0.f,0.f,0.f,0.f,0.f,0.f,0.f,0.f};
    v8f acc1 = {0.f,0.f,0.f,0.f,0.f,0.f,0.f,0.f};

    const int mrow  = wm * 16 + (lane & 15);
    const int khalf = lane >> 4;

    // ---- stage A tile for k-step kk into LDS buffer `buf` (async direct-to-LDS when available)
    auto stageA = [&](int kk, int buf) {
        const int r = kk / ksC, ci0 = (kk - r * ksC) << 5;
        const int ci = ci0 + ach * 8;
        #pragma unroll
        for (int si = 0; si < 3; ++si) {
            const unsigned short* src =
                wr + ((size_t)((r * 3 + si) * Cout + co0 + am)) * Cin + ci;
            unsigned short* dst = &As[buf][si][am][ach * 8];
            if (ci + 8 <= Cin) {
#if USE_ASYNC_LDS
                __builtin_amdgcn_global_load_async_to_lds_b128(
                    (__attribute__((address_space(1))) v4i32*)(void*)src,
                    (__attribute__((address_space(3))) v4i32*)(void*)dst, 0, 0);
#else
                *(u16x8*)dst = *(const u16x8*)src;
#endif
            } else {
                u16x8 v;
                #pragma unroll
                for (int j = 0; j < 8; ++j)
                    v[j] = (ci + j < Cin) ? src[j] : (unsigned short)0;
                *(u16x8*)dst = v;
            }
        }
    };

    // ---- load B tile (transposed staging: global -> regs), separate store (regs -> LDS)
    auto loadB = [&](int kk, unsigned short (&vb)[2][8]) {
        const int r = kk / ksC, ci0 = (kk - r * ksC) << 5;
        const int yin = y + r - 1;
        const bool yok = (yin >= 0) && (yin < H);
        #pragma unroll
        for (int q = 0; q < 2; ++q) {
            const int e = tid + q * 256;
            if (e < 320) {
                const int k = e / 10, ch = e - k * 10;
                const int ci = ci0 + k;
                const int xs = x0 - 8 + ch * 8;
                const bool cok = yok && (ci < Cin);
                const size_t rowbase = ((size_t)(n * Cin + ci) * H + (size_t)yin) * W;
                if (cok && xs >= 0 && xs + 8 <= W) {
                    u16x4 lo = *(const u16x4*)(in + rowbase + xs);
                    u16x4 hi = *(const u16x4*)(in + rowbase + xs + 4);
                    #pragma unroll
                    for (int j = 0; j < 4; ++j) { vb[q][j] = lo[j]; vb[q][j + 4] = hi[j]; }
                } else {
                    #pragma unroll
                    for (int j = 0; j < 8; ++j) {
                        const int x = xs + j;
                        vb[q][j] = (cok && x >= 0 && x < W)
                                       ? in[rowbase + x] : (unsigned short)0;
                    }
                }
            }
        }
    };
    auto storeB = [&](const unsigned short (&vb)[2][8], int buf) {
        #pragma unroll
        for (int q = 0; q < 2; ++q) {
            const int e = tid + q * 256;
            if (e < 320) {
                const int k = e / 10, ch = e - k * 10;
                #pragma unroll
                for (int j = 0; j < 8; ++j) Bs[buf][ch * 8 + j][k] = vb[q][j];
            }
        }
    };

    // ---- 3 column taps x 2 N-subtiles of WMMA from LDS buffer `buf`
    auto compute = [&](int buf) {
        #pragma unroll
        for (int s = 0; s < 3; ++s) {
            bfvec a, b0, b1;
            a.h[0] = *(const u16x8*)&As[buf][s][mrow][khalf * 16];
            a.h[1] = *(const u16x8*)&As[buf][s][mrow][khalf * 16 + 8];
            const int p0 = wn * 32 + (lane & 15) + s + 7;   // x = x0-8+p
            b0.h[0] = *(const u16x8*)&Bs[buf][p0][khalf * 16];
            b0.h[1] = *(const u16x8*)&Bs[buf][p0][khalf * 16 + 8];
            b1.h[0] = *(const u16x8*)&Bs[buf][p0 + 16][khalf * 16];
            b1.h[1] = *(const u16x8*)&Bs[buf][p0 + 16][khalf * 16 + 8];
            acc0 = __builtin_amdgcn_wmma_f32_16x16x32_bf16(
                       false, a.v, false, b0.v, (short)0, acc0, false, false);
            acc1 = __builtin_amdgcn_wmma_f32_16x16x32_bf16(
                       false, a.v, false, b1.v, (short)0, acc1, false, false);
        }
    };

    // ---- prologue: stage k-step 0 into buffer 0
    {
        unsigned short vb[2][8];
        stageA(0, 0);
        loadB(0, vb);
        storeB(vb, 0);
#if USE_ASYNC_LDS
        __builtin_amdgcn_s_wait_asynccnt(0);
#endif
    }
    __syncthreads();

    // ---- main pipeline: stage kk+1 while computing kk, one barrier per k-step
    for (int kk = 0; kk < K; ++kk) {
        const int cur = kk & 1;
        const bool more = (kk + 1) < K;
        unsigned short vb[2][8];
        if (more) {
            stageA(kk + 1, cur ^ 1);     // async: fire-and-forget into LDS
            loadB(kk + 1, vb);           // global -> regs (latency hidden by WMMAs)
        }
        compute(cur);
        if (more) {
            storeB(vb, cur ^ 1);
#if USE_ASYNC_LDS
            __builtin_amdgcn_s_wait_asynccnt(0);
#endif
        }
        __syncthreads();
    }

    // ---- epilogue: scale/bias + ReLU -> bf16 (C/D layout: VGPR v -> M=v+8*(lane>=16), N=lane%16)
    const int xbase = x0 + wn * 32 + (lane & 15);
    #pragma unroll
    for (int vi = 0; vi < 8; ++vi) {
        const int co = co0 + wm * 16 + vi + ((lane >> 4) << 3);
        const float sc = scale[co], bi = bias[co];
        const size_t rowbase = ((size_t)(n * Cout + co) * H + (size_t)y) * W;
        if (xbase < W) {
            float v = fmaxf(acc0[vi] * sc + bi, 0.f);
            out[rowbase + xbase] = f2bf(v);
        }
        if (xbase + 16 < W) {
            float v = fmaxf(acc1[vi] * sc + bi, 0.f);
            out[rowbase + xbase + 16] = f2bf(v);
        }
    }
}

// ---------------------------------------------------------------- 2x2 maxpool (bf16 in, bf16 + f32 out)
__global__ void maxpool2x2(const unsigned short* __restrict__ in,
                           unsigned short* __restrict__ obf, float* __restrict__ of32,
                           int C, int H, int W) {
    const int Ho = H >> 1, Wo = W >> 1;
    const long total = (long)32 * C * Ho * Wo;
    long i = (long)blockIdx.x * blockDim.x + threadIdx.x;
    if (i >= total) return;
    int xo = (int)(i % Wo); long t = i / Wo;
    int yo = (int)(t % Ho); t /= Ho;
    int c  = (int)(t % C);  int nb = (int)(t / C);
    size_t base = ((size_t)(nb * C + c) * H + (size_t)(yo * 2)) * W + (size_t)(xo * 2);
    float m = fmaxf(fmaxf(bf2f(in[base]),     bf2f(in[base + 1])),
                    fmaxf(bf2f(in[base + W]), bf2f(in[base + W + 1])));
    obf[i]  = f2bf(m);
    of32[i] = m;
}

// ---------------------------------------------------------------- host
extern "C" void kernel_launch(void* const* d_in, const int* in_sizes, int n_in,
                              void* d_out, int out_size, void* d_ws, size_t ws_size,
                              hipStream_t stream) {
    static const int LC[8][4] = {   // cin, cout, h, w (input dims of each conv)
        {3, 64, 224, 224}, {64, 64, 224, 224},
        {64, 128, 112, 112}, {128, 128, 112, 112},
        {128, 256, 56, 56}, {256, 256, 56, 56},
        {256, 512, 28, 28}, {512, 512, 28, 28}};
    const int N = 32;

    char* ws = (char*)d_ws;
    size_t off = 0;
    auto alloc = [&](size_t bytes) { off = (off + 255) & ~(size_t)255; size_t o = off; off += bytes; return o; };

    const size_t actBytes = (size_t)N * 64 * 224 * 224 * 2;   // largest bf16 activation
    size_t bufA = alloc(actBytes), bufB = alloc(actBytes);
    size_t wrOff[8], scOff[8], biOff[8];
    for (int l = 0; l < 8; ++l) {
        wrOff[l] = alloc((size_t)LC[l][0] * LC[l][1] * 9 * 2);
        scOff[l] = alloc((size_t)LC[l][1] * 4);
        biOff[l] = alloc((size_t)LC[l][1] * 4);
    }
    (void)ws_size; (void)n_in; (void)in_sizes; (void)out_size;

    // ---- parameter prep
    for (int l = 0; l < 8; ++l) {
        const float* Wt = (const float*)d_in[1 + 6 * l + 0];
        const float* b  = (const float*)d_in[1 + 6 * l + 1];
        const float* g  = (const float*)d_in[1 + 6 * l + 2];
        const float* be = (const float*)d_in[1 + 6 * l + 3];
        const float* mn = (const float*)d_in[1 + 6 * l + 4];
        const float* vr = (const float*)d_in[1 + 6 * l + 5];
        long wtot = (long)LC[l][0] * LC[l][1] * 9;
        pack_weights<<<(unsigned)((wtot + 255) / 256), 256, 0, stream>>>(
            Wt, (unsigned short*)(ws + wrOff[l]), LC[l][1], LC[l][0]);
        fold_bn<<<(LC[l][1] + 255) / 256, 256, 0, stream>>>(
            b, g, be, mn, vr, (float*)(ws + scOff[l]), (float*)(ws + biOff[l]), LC[l][1]);
    }

    // ---- input fp32 -> bf16
    {
        long n = (long)N * 3 * 224 * 224;
        cvt_f32_bf16<<<(unsigned)((n + 255) / 256), 256, 0, stream>>>(
            (const float*)d_in[0], (unsigned short*)(ws + bufA), n);
    }

    const size_t featElems[4] = {25690112ULL, 12845056ULL, 6422528ULL, 3211264ULL};
    size_t outOff = 0;
    size_t cur = bufA, alt = bufB;
    for (int blk = 0; blk < 4; ++blk) {
        for (int li = 0; li < 2; ++li) {
            int l = blk * 2 + li;
            int cin = LC[l][0], cout = LC[l][1], h = LC[l][2], w = LC[l][3];
            dim3 grid((w + 63) / 64, h, N * (cout / 64));
            size_t src = (li == 0) ? cur : alt;
            size_t dst = (li == 0) ? alt : cur;
            conv_bn_relu_wmma<<<grid, 256, 0, stream>>>(
                (const unsigned short*)(ws + src), (const unsigned short*)(ws + wrOff[l]),
                (const float*)(ws + scOff[l]), (const float*)(ws + biOff[l]),
                (unsigned short*)(ws + dst), cin, cout, h, w);
        }
        int l = blk * 2 + 1;
        int C = LC[l][1], H = LC[l][2], W = LC[l][3];
        long total = (long)N * C * (H / 2) * (W / 2);
        maxpool2x2<<<(unsigned)((total + 255) / 256), 256, 0, stream>>>(
            (const unsigned short*)(ws + cur), (unsigned short*)(ws + alt),
            (float*)d_out + outOff, C, H, W);
        outOff += featElems[blk];
        size_t tmp = cur; cur = alt; alt = tmp;   // pooled result now in `cur`
    }
}
